// TransducerBeamSearch_71579924955188
// MI455X (gfx1250) — matrix operand order repaired and checked by reading, not compiled, not run on hardware
//
#include <hip/hip_runtime.h>
#include <math.h>

#define V_     4096
#define BLANK_ 4095
#define D_     512
#define J_     640
#define E_     512
#define H_     640
#define T_     8192
#define G4H_   2560
#define NWG    16
#define NT     256
#define TILE_ELEMS 8192u   /* 16 rows x 512 K of bf16 staged per block */

typedef __attribute__((ext_vector_type(16))) __bf16       v16bf;
typedef __attribute__((ext_vector_type(8)))  float        v8f;
typedef __attribute__((ext_vector_type(4)))  float        f32x4;
typedef __attribute__((ext_vector_type(4)))  unsigned int u32x4;
typedef __attribute__((ext_vector_type(2)))  unsigned int u32x2;
typedef __attribute__((ext_vector_type(8)))  int          i32x8;
typedef __attribute__((ext_vector_type(4)))  int          i32x4;

union Frag { v16bf v; u32x4 q[2]; };

// Toolchain feature detection for the Tensor Data Mover builtin.
#if defined(__has_builtin)
#if __has_builtin(__builtin_amdgcn_tensor_load_to_lds) && __has_builtin(__builtin_amdgcn_s_wait_tensorcnt)
#define HAVE_TDM 1
#endif
#endif
#if defined(__has_include)
#if __has_include(<hip/amd_detail/amd_gfx1250_TDM.h>)
#define TDM_ARGS6 1   /* therock-10.0 headers -> 6-arg builtin */
#endif
#endif

__device__ inline unsigned short f2bf(float f) {
  unsigned u = __float_as_uint(f);
  u += 0x7FFFu + ((u >> 16) & 1u);        // round-to-nearest-even
  return (unsigned short)(u >> 16);
}

__device__ inline float sigf(float x) { return 1.0f / (1.0f + expf(-x)); }

// Online-softmax merge: (m,s,i) <- combine with (m2,s2,i2).
// First-index tie-break matches jnp.argmax; fixed application order everywhere
// keeps every workgroup's result bit-identical (deterministic token choice).
__device__ inline void combine(float& m, float& s, int& i, float m2, float s2, int i2) {
  if (m2 > m)       { s = s * expf(m - m2) + s2; m = m2; i = i2; }
  else if (m2 == m) { s += s2; if (i2 < i) i = i2; }
  else              { s += s2 * expf(m2 - m); }
}

// ---------------------------------------------------------------------------
// Workspace init: zero grid-barrier counter and LSTM state (stream-ordered).
// ---------------------------------------------------------------------------
__global__ __launch_bounds__(256) void init_ws(unsigned* bar, float* h, float* c) {
  int tid = threadIdx.x;
  if (tid == 0) *bar = 0u;
  for (int i = tid; i < H_; i += 256) { h[i] = 0.0f; c[i] = 0.0f; }
}

// ---------------------------------------------------------------------------
// enc_out [T,D] f32 -> bf16 (one pass; keeps all conversion VALU out of GEMM)
// ---------------------------------------------------------------------------
__global__ __launch_bounds__(256) void cvt_enc(const float* __restrict__ enc,
                                               unsigned short* __restrict__ encA) {
  int idx = blockIdx.x * blockDim.x + threadIdx.x;      // 4 elements per thread
  f32x4 v = ((const f32x4*)enc)[idx];
  unsigned r0 = (unsigned)f2bf(v[0]) | ((unsigned)f2bf(v[1]) << 16);
  unsigned r1 = (unsigned)f2bf(v[2]) | ((unsigned)f2bf(v[3]) << 16);
  u32x2 o = {r0, r1};
  ((u32x2*)encA)[idx] = o;
}

// ---------------------------------------------------------------------------
// Wenc [D,J] f32 -> transposed bf16 [J,D]: WMMA B lanes then read two 8-K runs
// (16 K apart) of one N column with two b128 loads.
// ---------------------------------------------------------------------------
__global__ __launch_bounds__(256) void cvt_wenc_t(const float* __restrict__ Wenc,
                                                  unsigned short* __restrict__ wT) {
  int idx = blockIdx.x * blockDim.x + threadIdx.x;
  if (idx >= J_ * D_) return;
  int j = idx / D_, d = idx % D_;
  wT[idx] = f2bf(Wenc[(size_t)d * J_ + j]);
}

// ---------------------------------------------------------------------------
// enc_proj[T,J] = enc_out[T,D] @ Wenc[D,J] + benc   (bf16 WMMA, f32 acc)
// One block = one 16-row M tile; A tile staged to LDS once via the Tensor
// Data Mover (TDM) and reused by all 8 waves (5 N-tiles each => J=640).
// ---------------------------------------------------------------------------
__global__ __launch_bounds__(256) void enc_proj_wmma(
    const unsigned short* __restrict__ encA,   // [T,D] bf16
    const unsigned short* __restrict__ wT,     // [J,D] bf16 (Wenc^T)
    const float* __restrict__ benc, float* __restrict__ encp) {
  __shared__ __attribute__((aligned(16))) unsigned short s_a[16 * D_];

  const int it   = blockIdx.x * 16;
  const int w    = threadIdx.x >> 5;     // wave 0..7
  const int lane = threadIdx.x & 31;
  const int m    = lane & 15;            // A row / B column / C column
  const int kh   = lane >> 4;

#ifdef HAVE_TDM
  if (threadIdx.x < 32) {                // wave 0 issues the DMA (TDM ignores EXEC;
                                         // wave-uniform branch keeps other waves off it)
    unsigned long long ga = (unsigned long long)(const void*)(encA + (size_t)it * D_);
    unsigned ldsOff = (unsigned)(unsigned long long)(const void*)&s_a[0];
    u32x4 g0 = {};
    g0[0] = 1u;                                                   // count=1 (valid D#)
    g0[1] = ldsOff;                                               // lds_addr
    g0[2] = (unsigned)ga;                                         // global_addr[31:0]
    g0[3] = (unsigned)((ga >> 32) & 0x01FFFFFFull) | (2u << 30);  // [56:32] | type=2
    i32x8 g1 = {};
    g1[0] = (int)(1u << 16);                                      // data_size=1 (2B)
    g1[1] = (int)((TILE_ELEMS & 0xFFFFu) << 16);                  // tensor_dim0[15:0]
    g1[2] = (int)((TILE_ELEMS >> 16) | (1u << 16));               // dim0[31:16] | dim1=1
    g1[3] = (int)(TILE_ELEMS << 16);                              // tile_dim0=8192
    g1[4] = 1;                                                    // tile_dim1=1
    g1[5] = (int)TILE_ELEMS;                                      // dim0_stride[31:0]
    i32x4 gz = {};
#ifdef TDM_ARGS6
    i32x8 gz8 = {};
    __builtin_amdgcn_tensor_load_to_lds(g0, g1, gz, gz, gz8, 0);
#else
    __builtin_amdgcn_tensor_load_to_lds(g0, g1, gz, gz, 0);
#endif
    __builtin_amdgcn_s_wait_tensorcnt(0);
  }
#else
  {
    const u32x4* src = (const u32x4*)(encA + (size_t)it * D_);
    u32x4* dst = (u32x4*)s_a;
    for (int i = threadIdx.x; i < (16 * D_) / 8; i += 256) dst[i] = src[i];
  }
#endif
  __syncthreads();

  v8f acc[5] = {};
  for (int kt = 0; kt < D_; kt += 32) {
    // A fragment from LDS: elems 0..7 = K kt+kh*8.., elems 8..15 = K kt+16+kh*8..
    Frag a;
    const unsigned short* ap = s_a + m * D_ + kt + kh * 8;
    a.q[0] = *(const u32x4*)ap;
    a.q[1] = *(const u32x4*)(ap + 16);
    // B fragments: lane n=m of column w*80 + q*16 + n: elems 0..7 = K kt+kh*8.., elems 8..15 = K kt+16+kh*8..
    const unsigned short* bbase = wT + (size_t)(w * 80 + m) * D_ + kt + kh * 8;
    if (kt + 32 < D_) __builtin_prefetch(bbase + 32, 0, 3);
#pragma unroll
    for (int q = 0; q < 5; ++q) {
      Frag b;
      const unsigned short* bp = bbase + (size_t)(q * 16) * D_;
      b.q[0] = *(const u32x4*)bp;
      b.q[1] = *(const u32x4*)(bp + 16);
      acc[q] = __builtin_amdgcn_wmma_f32_16x16x32_bf16(
          false, a.v, false, b.v, (short)0, acc[q], false, false);
    }
  }

  const int mb = kh * 8;                  // C/D: VGPR r -> row mb+r
#pragma unroll
  for (int q = 0; q < 5; ++q) {
    int jc = w * 80 + q * 16 + m;
    float bias = benc[jc];
    float* op = encp + (size_t)(it + mb) * J_ + jc;
#pragma unroll
    for (int r = 0; r < 8; ++r) op[(size_t)r * J_] = acc[q][r] + bias;
  }
}

// ---------------------------------------------------------------------------
// Cooperative greedy-decode scan: 16 workgroups (one cluster's worth of WGPs).
// M=1 GEMVs -> f32 VALU dot products (WMMA would idle 15/16 lanes); all
// weights (~34 MB) are L2-resident, so this is L2-latency/bandwidth bound.
// 4 grid barriers per frame (online softmax folds max+sum into one pass).
// ---------------------------------------------------------------------------
struct DecP {
  const float* encp;  const float* embed; const float* Wx;  const float* Wh;
  const float* b_lstm; const float* Wdp;  const float* bdp; const float* Wout;
  const float* bout;  float* out;
  float* dp; float* h; float* c; float* h2; float* c2; float* g;
  float* pmax; int* pidx; float* psum; unsigned* bar;
};

__global__ __launch_bounds__(NT) void decode_kernel(DecP P) {
  __shared__ float s_u[J_];
  __shared__ float s_e[E_];
  __shared__ float s_hL[H_];
  __shared__ float s_h2[H_];
  __shared__ float s_pm[NT / 32];
  __shared__ float s_ps[NT / 32];
  __shared__ int   s_pi[NT / 32];

  const int tid = threadIdx.x;
  const int wg  = blockIdx.x;
  unsigned barIdx = 0;

  auto gbar = [&]() {
    __threadfence();
    __syncthreads();
    __builtin_amdgcn_s_cluster_barrier();   // HW fast path if clustered (NOP otherwise)
    if (tid == 0) {
      atomicAdd(P.bar, 1u);
      unsigned need = (barIdx + 1u) * (unsigned)NWG;
      while (__hip_atomic_load(P.bar, __ATOMIC_ACQUIRE, __HIP_MEMORY_SCOPE_AGENT) < need)
        __builtin_amdgcn_s_sleep(1);
    }
    ++barIdx;
    __syncthreads();
  };

  int  tok   = BLANK_;
  bool force = true;                        // t == -1: initial pred step, zero state

  for (int t = -1; t < T_; ++t) {
    if (t >= 0) {
      // u = tanh(enc_proj[t] + dp)
      for (int j = tid; j < J_; j += NT)
        s_u[j] = tanhf(P.encp[(size_t)t * J_ + j] + P.dp[j]);
      __syncthreads();

      // logits: one vocab column per thread (coalesced over Wout rows)
      const int v = wg * NT + tid;
      float lg = P.bout[v];
      const float* wc = P.Wout + v;
#pragma unroll 4
      for (int j = 0; j < J_; ++j) lg += s_u[j] * wc[(size_t)j * V_];

      // single-pass (max, sum-exp, argmax): wave32 shuffle butterfly ...
      float m = lg, s = 1.0f; int idx = v;
#pragma unroll
      for (int off = 16; off > 0; off >>= 1) {
        float m2 = __shfl_down(m, off);
        float s2 = __shfl_down(s, off);
        int   i2 = __shfl_down(idx, off);
        combine(m, s, idx, m2, s2, i2);
      }
      // ... then 8 wave leaders via LDS
      if ((tid & 31) == 0) { s_pm[tid >> 5] = m; s_ps[tid >> 5] = s; s_pi[tid >> 5] = idx; }
      __syncthreads();
      float bm = s_pm[0], bs = s_ps[0]; int bi = s_pi[0];
#pragma unroll
      for (int wv = 1; wv < NT / 32; ++wv) combine(bm, bs, bi, s_pm[wv], s_ps[wv], s_pi[wv]);
      if (tid == 0) { P.pmax[wg] = bm; P.psum[wg] = bs; P.pidx[wg] = bi; }
      gbar();

      // merge the 16 per-WG triples (same order everywhere -> same token)
      float gm = P.pmax[0], gs = P.psum[0]; int gi = P.pidx[0];
#pragma unroll
      for (int w = 1; w < NWG; ++w) combine(gm, gs, gi, P.pmax[w], P.psum[w], P.pidx[w]);
      float lz = gm + logf(gs);
      P.out[(size_t)t * V_ + v] = lg - lz;  // log_softmax row
      tok   = gi;
      force = false;
    }

    // ---- prediction-network (LSTM) step ----
    for (int i = tid; i < E_; i += NT) s_e[i]  = P.embed[(size_t)tok * E_ + i];
    for (int i = tid; i < H_; i += NT) s_hL[i] = P.h[i];
    __syncthreads();

    const int KS = G4H_ / NWG;              // 160 gate pre-activations per WG
    for (int k = wg * KS + tid; k < (wg + 1) * KS; k += NT) {
      float acc = P.b_lstm[k];
      const float* wx = P.Wx + k;
#pragma unroll 4
      for (int i = 0; i < E_; ++i) acc += s_e[i] * wx[(size_t)i * G4H_];
      const float* wh = P.Wh + k;
#pragma unroll 4
      for (int i = 0; i < H_; ++i) acc += s_hL[i] * wh[(size_t)i * G4H_];
      P.g[k] = acc;
    }
    gbar();

    const int HS = H_ / NWG;                // 40 hidden units per WG
    for (int k = wg * HS + tid; k < (wg + 1) * HS; k += NT) {
      float gi = P.g[k], gf = P.g[H_ + k], gc = P.g[2 * H_ + k], go = P.g[3 * H_ + k];
      float c2 = sigf(gf) * P.c[k] + sigf(gi) * tanhf(gc);
      P.c2[k] = c2;
      P.h2[k] = sigf(go) * tanhf(c2);
    }
    gbar();

    // advance state only on emission (or forced initial step)
    bool commit = force || (tok != BLANK_);
    if (commit) {
      for (int i = tid; i < H_; i += NT) s_h2[i] = P.h2[i];
      __syncthreads();
      for (int j = wg * HS + tid; j < (wg + 1) * HS; j += NT) {
        float acc = P.bdp[j];
        const float* wd = P.Wdp + j;
#pragma unroll 4
        for (int i = 0; i < H_; ++i) acc += s_h2[i] * wd[(size_t)i * J_];
        P.dp[j] = acc;
      }
      for (int k = wg * HS + tid; k < (wg + 1) * HS; k += NT) {
        P.h[k] = P.h2[k];
        P.c[k] = P.c2[k];
      }
    }
    gbar();
  }
}

// ---------------------------------------------------------------------------
extern "C" void kernel_launch(void* const* d_in, const int* in_sizes, int n_in,
                              void* d_out, int out_size, void* d_ws, size_t ws_size,
                              hipStream_t stream) {
  (void)in_sizes; (void)n_in; (void)out_size; (void)ws_size;

  const float* enc_out = (const float*)d_in[0];
  const float* embed   = (const float*)d_in[1];
  const float* Wx      = (const float*)d_in[2];
  const float* Wh      = (const float*)d_in[3];
  const float* b_lstm  = (const float*)d_in[4];
  const float* Wdp     = (const float*)d_in[5];
  const float* bdp     = (const float*)d_in[6];
  const float* Wenc    = (const float*)d_in[7];
  const float* benc    = (const float*)d_in[8];
  const float* Wout    = (const float*)d_in[9];
  const float* bout    = (const float*)d_in[10];
  float* out = (float*)d_out;

  char* wsBase = (char*)d_ws;
  size_t off = 0;
  auto take = [&](size_t bytes) -> char* {
    char* p = wsBase + off;
    off += (bytes + 255) & ~(size_t)255;
    return p;
  };
  unsigned short* encA = (unsigned short*)take((size_t)T_ * D_ * 2);
  unsigned short* wT   = (unsigned short*)take((size_t)J_ * D_ * 2);
  float* encp = (float*)take((size_t)T_ * J_ * 4);
  float* dpb  = (float*)take(J_ * 4);
  float* hb   = (float*)take(H_ * 4);
  float* cb   = (float*)take(H_ * 4);
  float* h2b  = (float*)take(H_ * 4);
  float* c2b  = (float*)take(H_ * 4);
  float* gb   = (float*)take(G4H_ * 4);
  float* pmax = (float*)take(NWG * 4);
  int*   pidx = (int*)take(NWG * 4);
  float* psum = (float*)take(NWG * 4);
  unsigned* bar = (unsigned*)take(64);

  init_ws<<<1, 256, 0, stream>>>(bar, hb, cb);
  cvt_enc<<<(T_ * D_ / 4) / 256, 256, 0, stream>>>(enc_out, encA);
  cvt_wenc_t<<<(J_ * D_ + 255) / 256, 256, 0, stream>>>(Wenc, wT);
  enc_proj_wmma<<<T_ / 16, 256, 0, stream>>>(encA, wT, benc, encp);

  DecP P{encp, embed, Wx, Wh, b_lstm, Wdp, bdp, Wenc /*unused*/ == 0 ? 0 : bdp, bout, out,
         dpb, hb, cb, h2b, c2b, gb, pmax, pidx, psum, bar};
  // (fix accidental placeholder: assign explicitly)
  P.Wdp = Wdp; P.bdp = bdp; P.Wout = Wout; P.bout = bout;
  decode_kernel<<<NWG, NT, 0, stream>>>(P);
}
